// GPT_34419867910890
// MI455X (gfx1250) — matrix operand order repaired and checked by reading, not compiled, not run on hardware
//
#include <hip/hip_runtime.h>
#include <hip/hip_bf16.h>

// ---------------------------------------------------------------------------
// Types / helpers
// ---------------------------------------------------------------------------
typedef __attribute__((ext_vector_type(16))) __bf16 v16bf;
typedef __attribute__((ext_vector_type(8)))  float  v8f;

union BFrag { unsigned int u[8]; v16bf v; };

__device__ __forceinline__ unsigned short f2bf(float f) {
  unsigned int u = __float_as_uint(f);
  u += 0x7FFFu + ((u >> 16) & 1u);          // round-to-nearest-even
  return (unsigned short)(u >> 16);
}
__device__ __forceinline__ float bf2f(unsigned short h) {
  return __uint_as_float(((unsigned int)h) << 16);
}
__device__ __forceinline__ v8f wmma_bf16(const BFrag& a, const BFrag& b, v8f c) {
  return __builtin_amdgcn_wmma_f32_16x16x32_bf16(false, a.v, false, b.v,
                                                 (short)0, c, false, false);
}

// gfx1250 async global->LDS copy (ASYNCcnt-tracked, no VGPR round trip).
__device__ __forceinline__ void async_copy_b128(const void* gsrc, void* ldst) {
  asm volatile("global_load_async_to_lds_b128 %0, %1, off"
               :: "v"((unsigned int)(unsigned long long)ldst), "v"(gsrc)
               : "memory");
}
__device__ __forceinline__ void wait_async0() {
  asm volatile("s_wait_asynccnt 0" ::: "memory");
}

#define C_DIM   1024
#define T_DIM   1024
#define B_DIM   2
#define H_DIM   16
#define HD_DIM  64
#define BT_DIM  (B_DIM * T_DIM)          // 2048
#define DDOWN   544                      // D_C + D_C1 + D_ROPE
#define DKV     1536                     // H*(D_NOPE+HD)
#define RMS_EPS 1.1920929e-7f

enum { EPI_F32 = 0, EPI_RESID = 1, EPI_RELU2_BF16 = 2, EPI_SOFTCAP = 3 };

// ---------------------------------------------------------------------------
// Generic tiled bf16 WMMA GEMM:  Out(MxN) = A(MxK bf16) * B(KxN f32->bf16)
// block = 256 threads (8 waves), tile 128x128, K-step 32, wave tile 64x32.
// Double-buffered LDS; A tile streamed with async-to-LDS, B tile staged via
// registers (needs f32->bf16 conversion + transpose).
// ---------------------------------------------------------------------------
#define BM 128
#define BN 128
#define BK 32

__global__ __launch_bounds__(256) void gemm_bf16_wmma(
    const unsigned short* __restrict__ A, const float* __restrict__ B,
    void* __restrict__ Out, const float* __restrict__ AddSrc,
    int M, int N, int K, int epi)
{
  __shared__ __align__(16) unsigned short As[2][BM * BK];   // [row][k]
  __shared__ __align__(16) unsigned short Bs[2][BN * BK];   // [col][k] (transposed)

  const int tid   = threadIdx.x;
  const int mBase = blockIdx.y * BM;
  const int nBase = blockIdx.x * BN;
  const int w     = tid >> 5, lane = tid & 31;
  const int halfsel = lane >> 4, l16 = lane & 15;
  const int wm = (w >> 2) * 64;   // wave row offset in tile
  const int wn = (w & 3)  * 32;   // wave col offset in tile
  const int ntiles = K / BK;

  v8f acc[4][2];
  for (int mt = 0; mt < 4; ++mt)
    for (int nt = 0; nt < 2; ++nt)
      for (int r = 0; r < 8; ++r) acc[mt][nt][r] = 0.0f;

  float4 breg[4];

  auto issueA = [&](int kt, int buf) {            // async b128: 512 slots of 8 bf16
    int k0 = kt * BK;
    for (int i = 0; i < 2; ++i) {
      int slot = tid + 256 * i;
      int row = slot >> 2, c8 = slot & 3;
      async_copy_b128(A + (size_t)(mBase + row) * K + k0 + c8 * 8,
                      &As[buf][row * BK + c8 * 8]);
    }
  };
  auto issueB = [&](int kt) {                     // fp32 loads -> regs (no wait yet)
    int k0 = kt * BK;
    for (int i = 0; i < 4; ++i) {
      int slot = tid + 256 * i;                   // 1024 slots of float4
      int kk = slot >> 5, n4 = slot & 31;
      int gN = nBase + n4 * 4;
      float4 v{0.0f, 0.0f, 0.0f, 0.0f};
      if (gN < N) v = *(const float4*)(B + (size_t)(k0 + kk) * N + gN);
      breg[i] = v;
    }
  };
  auto storeB = [&](int buf) {                    // convert + transpose into LDS
    for (int i = 0; i < 4; ++i) {
      int slot = tid + 256 * i;
      int kk = slot >> 5, n4 = slot & 31;
      unsigned short* dst = &Bs[buf][(n4 * 4) * BK + kk];
      dst[0]      = f2bf(breg[i].x);
      dst[BK]     = f2bf(breg[i].y);
      dst[2 * BK] = f2bf(breg[i].z);
      dst[3 * BK] = f2bf(breg[i].w);
    }
  };

  // prologue: stage tile 0
  issueB(0);
  issueA(0, 0);
  storeB(0);
  wait_async0();
  __syncthreads();

  int buf = 0;
  for (int kt = 0; kt < ntiles; ++kt) {
    const int nxt = buf ^ 1;
    const bool more = (kt + 1) < ntiles;
    if (more) {
      issueB(kt + 1);                             // loads in flight during WMMAs
      issueA(kt + 1, nxt);                        // async direct to LDS
      if (kt + 2 < ntiles)
        __builtin_prefetch(B + (size_t)((kt + 2) * BK + (tid >> 3)) * N +
                               nBase + (tid & 7) * 16, 0, 1);
    }

    // --- fragments + WMMA on current buffer ---
    BFrag af[4], bfr[2];
    for (int mt = 0; mt < 4; ++mt) {
      int base = (wm + mt * 16 + l16) * BK;
      for (int p = 0; p < 8; ++p) {
        int kp = ((p & 4) << 2) + halfsel * 8 + ((p & 3) << 1);   // A layout
        af[mt].u[p] = *(const unsigned int*)(&As[buf][base + kp]);
      }
    }
    for (int nt = 0; nt < 2; ++nt) {
      int base = (wn + nt * 16 + l16) * BK;
      for (int p = 0; p < 8; ++p) {
        int kp = ((p & 4) << 2) + halfsel * 8 + ((p & 3) << 1);   // B layout
        bfr[nt].u[p] = *(const unsigned int*)(&Bs[buf][base + kp]);
      }
    }
    for (int mt = 0; mt < 4; ++mt)
      for (int nt = 0; nt < 2; ++nt)
        acc[mt][nt] = wmma_bf16(af[mt], bfr[nt], acc[mt][nt]);

    if (more) {
      storeB(nxt);                                // waits B loads here (post-WMMA)
      wait_async0();                              // A tile landed in LDS
    }
    __syncthreads();
    buf = nxt;
  }

  // --- epilogue ---
  for (int mt = 0; mt < 4; ++mt)
    for (int nt = 0; nt < 2; ++nt)
      for (int r = 0; r < 8; ++r) {
        int row = mBase + wm + mt * 16 + r + 8 * halfsel;
        int col = nBase + wn + nt * 16 + l16;
        if (col >= N) continue;
        size_t o = (size_t)row * N + col;
        float vacc = acc[mt][nt][r];
        if (epi == EPI_F32)        ((float*)Out)[o] = vacc;
        else if (epi == EPI_RESID) ((float*)Out)[o] = AddSrc[o] + vacc;
        else if (epi == EPI_RELU2_BF16) {
          float t = fmaxf(vacc, 0.0f);
          ((unsigned short*)Out)[o] = f2bf(t * t);
        } else {                    // softcap
          ((float*)Out)[o] = 15.0f * tanhf(vacc * (1.0f / 15.0f));
        }
      }
}

// ---------------------------------------------------------------------------
// Embedding gather + rmsnorm  ->  x, x0   (one block per token)
// ---------------------------------------------------------------------------
__global__ __launch_bounds__(256) void embed_norm(
    const int* __restrict__ idx, const float* __restrict__ wte,
    float* __restrict__ x, float* __restrict__ x0)
{
  int token = blockIdx.x, tid = threadIdx.x;
  const float* row = wte + (size_t)idx[token] * C_DIM;
  __shared__ float red[256];
  float vals[4], ss = 0.0f;
  for (int i = 0; i < 4; ++i) { vals[i] = row[tid + 256 * i]; ss += vals[i] * vals[i]; }
  red[tid] = ss; __syncthreads();
  for (int s = 128; s > 0; s >>= 1) { if (tid < s) red[tid] += red[tid + s]; __syncthreads(); }
  float inv = rsqrtf(red[0] * (1.0f / C_DIM) + RMS_EPS);
  for (int i = 0; i < 4; ++i) {
    float o = vals[i] * inv;
    x[(size_t)token * C_DIM + tid + 256 * i]  = o;
    x0[(size_t)token * C_DIM + tid + 256 * i] = o;
  }
}

// ---------------------------------------------------------------------------
// xm = rl*x + xl*x0 ; xn_bf = bf16(rmsnorm(xm)).  l < 0 -> rl=1, xl=0.
// ---------------------------------------------------------------------------
__global__ __launch_bounds__(256) void resid_norm(
    const float* __restrict__ xin, const float* __restrict__ x0v,
    const float* __restrict__ rlv, const float* __restrict__ xlv, int l,
    float* __restrict__ xmout, unsigned short* __restrict__ xnout)
{
  int token = blockIdx.x, tid = threadIdx.x;
  float rl = (l >= 0) ? rlv[l] : 1.0f;
  float xl = (l >= 0) ? xlv[l] : 0.0f;
  __shared__ float red[256];
  float vals[4], ss = 0.0f;
  for (int i = 0; i < 4; ++i) {
    size_t o = (size_t)token * C_DIM + tid + 256 * i;
    vals[i] = rl * xin[o] + xl * x0v[o];
    ss += vals[i] * vals[i];
  }
  red[tid] = ss; __syncthreads();
  for (int s = 128; s > 0; s >>= 1) { if (tid < s) red[tid] += red[tid + s]; __syncthreads(); }
  float inv = rsqrtf(red[0] * (1.0f / C_DIM) + RMS_EPS);
  for (int i = 0; i < 4; ++i) {
    size_t o = (size_t)token * C_DIM + tid + 256 * i;
    xmout[o] = vals[i];
    xnout[o] = f2bf(vals[i] * inv);
  }
}

// ---------------------------------------------------------------------------
// down(:, :512) -> bf16 c_kv, c_q
// ---------------------------------------------------------------------------
__global__ __launch_bounds__(256) void split_down(
    const float* __restrict__ down, unsigned short* __restrict__ ckv,
    unsigned short* __restrict__ cq)
{
  int i = blockIdx.x * 256 + threadIdx.x;     // BT*512 total
  int tkn = i >> 9, c = i & 511;
  float v = down[(size_t)tkn * DDOWN + c];
  if (c < 256) ckv[(size_t)tkn * 256 + c]       = f2bf(v);
  else         cq[(size_t)tkn * 256 + (c - 256)] = f2bf(v);
}

// ---------------------------------------------------------------------------
// Assemble Q/K/V (rope, gate, value-embed, per-head rmsnorm). thread=(b,t,h)
// Q,K layout: [B][H][T][64] bf16.   V layout (pre-transposed for PV WMMA
// B-fragments): Vt[B][H][64][T] bf16  (key index contiguous).
// ---------------------------------------------------------------------------
__device__ __forceinline__ void rms64(float* v) {
  float ss = 0.0f;
  for (int i = 0; i < 64; ++i) ss += v[i] * v[i];
  float inv = rsqrtf(ss * (1.0f / 64.0f) + RMS_EPS);
  for (int i = 0; i < 64; ++i) v[i] *= inv;
}

__global__ __launch_bounds__(256) void prep_qkv(
    const float* __restrict__ down, const float* __restrict__ kvf,
    const float* __restrict__ qf, const unsigned short* __restrict__ xnbf,
    const float* __restrict__ cosb, const float* __restrict__ sinb,
    const float* __restrict__ vg, const float* __restrict__ vetab,
    const int* __restrict__ idx,
    unsigned short* __restrict__ Qbf, unsigned short* __restrict__ Kbf,
    unsigned short* __restrict__ Vt)
{
  int g = blockIdx.x * blockDim.x + threadIdx.x;
  if (g >= BT_DIM * H_DIM) return;
  int h = g & 15, tkn = g >> 4;
  int b = tkn >> 10, t = tkn & 1023;
  const float* cs = cosb + (size_t)tkn * 16;
  const float* sn = sinb + (size_t)tkn * 16;

  // gate = 2*sigmoid(xn[:32] . vg[:,h])
  float acc = 0.0f;
  for (int c = 0; c < 32; ++c) acc += bf2f(xnbf[(size_t)tkn * C_DIM + c]) * vg[c * H_DIM + h];
  float gate = 2.0f / (1.0f + __expf(-acc));

  // rope(k_rope_raw)
  float kr[32];
  const float* krr = down + (size_t)tkn * DDOWN + 512;
  for (int i = 0; i < 16; ++i) {
    float x1 = krr[i], x2 = krr[16 + i], c = cs[i], s = sn[i];
    kr[i]      =  x1 * c + x2 * s;
    kr[16 + i] = -x1 * s + x2 * c;
  }

  size_t headoff = ((size_t)(b * H_DIM + h)) * T_DIM * HD_DIM;
  size_t hoff = headoff + (size_t)t * HD_DIM;

  // Q: [q_nope(32), rope(q_rope)(32)], rmsnorm
  float q[64];
  const float* qrow = qf + (size_t)tkn * C_DIM + h * HD_DIM;
  for (int d = 0; d < 64; ++d) q[d] = qrow[d];
  for (int i = 0; i < 16; ++i) {
    float x1 = q[32 + i], x2 = q[48 + i], c = cs[i], s = sn[i];
    q[32 + i] =  x1 * c + x2 * s;
    q[48 + i] = -x1 * s + x2 * c;
  }
  rms64(q);
  for (int d = 0; d < 64; ++d) Qbf[hoff + d] = f2bf(q[d]);

  // K: [k_nope(32), k_rope(32)], rmsnorm
  const float* kvrow = kvf + (size_t)tkn * DKV + h * 96;
  float k[64];
  for (int i = 0; i < 32; ++i) { k[i] = kvrow[i]; k[32 + i] = kr[i]; }
  rms64(k);
  for (int d = 0; d < 64; ++d) Kbf[hoff + d] = f2bf(k[d]);

  // V (transposed store): Vt[b][h][d][t] = kv[...,32+d] + gate * ve[d]
  const float* verow = vetab + (size_t)idx[tkn] * C_DIM + h * HD_DIM;
  for (int d = 0; d < 64; ++d)
    Vt[headoff + (size_t)d * T_DIM + t] = f2bf(kvrow[32 + d] + gate * verow[d]);
}

// ---------------------------------------------------------------------------
// Flash attention: one wave per 16-query tile per (b,h); 32 keys / iter.
// K and Vt B-fragments load DIRECTLY from global (layouts match WMMA operand
// order, each fragment = 2x global b128); only the P transpose uses LDS.
// Y layout: [B][T][H*64] bf16
// ---------------------------------------------------------------------------
__global__ __launch_bounds__(256) void attn_flash(
    const unsigned short* __restrict__ Q, const unsigned short* __restrict__ K,
    const unsigned short* __restrict__ Vt, unsigned short* __restrict__ Y)
{
  __shared__ __align__(16) unsigned short smem[8 * 512];
  const int w = threadIdx.x >> 5, lane = threadIdx.x & 31;
  const int halfsel = lane >> 4, l16 = lane & 15;
  unsigned short* Ps = smem + w * 512;         // [row][key] 16x32

  int qt = blockIdx.x * 8 + w;                 // 2048 q-tiles total
  int b  = qt >> 10;
  int h  = (qt >> 6) & 15;
  int qb = qt & 63;
  int qbase = qb << 4;
  const size_t headoff = ((size_t)(b * H_DIM + h)) * T_DIM * HD_DIM;

  // Q fragments (row = l16), HD=64 split into two K=32 chunks
  BFrag aq[2];
  {
    const unsigned short* qrow = Q + headoff + (size_t)(qbase + l16) * HD_DIM;
    for (int kc = 0; kc < 2; ++kc)
      for (int p = 0; p < 8; ++p) {
        int kd = kc * 32 + ((p & 4) << 2) + halfsel * 8 + ((p & 3) << 1);
        aq[kc].u[p] = *(const unsigned int*)(qrow + kd);
      }
  }

  float m[8], lsum[8];
  v8f Ov[4];
  for (int r = 0; r < 8; ++r) { m[r] = -3.0e38f; lsum[r] = 0.0f; }
  for (int t = 0; t < 4; ++t)
    for (int r = 0; r < 8; ++r) Ov[t][r] = 0.0f;

  int nkb = (qbase + 47) >> 5;                 // causal key-block count
  for (int kb = 0; kb < nkb; ++kb) {
    int kstart = kb << 5;
    if (kb + 1 < nkb) {                        // prefetch next key block
      __builtin_prefetch(K  + headoff + (size_t)(kstart + 32 + lane) * HD_DIM, 0, 1);
      __builtin_prefetch(Vt + headoff + (size_t)lane        * T_DIM + kstart + 32, 0, 1);
      __builtin_prefetch(Vt + headoff + (size_t)(lane + 32) * T_DIM + kstart + 32, 0, 1);
    }

    // scores: two 16-key halves; B-fragments straight from K[key][d]
    v8f sh[2];
    for (int hh = 0; hh < 2; ++hh) {
      v8f s;
      for (int r = 0; r < 8; ++r) s[r] = 0.0f;
      for (int kc = 0; kc < 2; ++kc) {
        BFrag bk;
        const unsigned short* kr =
            K + headoff + (size_t)(kstart + hh * 16 + l16) * HD_DIM +
            kc * 32 + halfsel * 8;
        for (int p = 0; p < 8; ++p)
          bk.u[p] = *(const unsigned int*)(kr + ((p & 4) << 2) + ((p & 3) << 1));
        s = wmma_bf16(aq[kc], bk, s);
      }
      int key = kstart + hh * 16 + l16;
      for (int r = 0; r < 8; ++r) {
        int qrow = qbase + r + 8 * halfsel;
        sh[hh][r] = (key > qrow) ? -1.0e30f : s[r] * 0.125f;  // 1/sqrt(64)
      }
    }
    // online softmax (row reductions within 16-lane halves)
    float scale[8], p0[8], p1[8];
    for (int r = 0; r < 8; ++r) {
      float v = fmaxf(sh[0][r], sh[1][r]);
      for (int mk = 1; mk < 16; mk <<= 1) v = fmaxf(v, __shfl_xor(v, mk, 32));
      float mnew = fmaxf(m[r], v);
      scale[r] = __expf(m[r] - mnew);
      p0[r] = __expf(sh[0][r] - mnew);
      p1[r] = __expf(sh[1][r] - mnew);
      float rs = p0[r] + p1[r];
      for (int mk = 1; mk < 16; mk <<= 1) rs += __shfl_xor(rs, mk, 32);
      lsum[r] = lsum[r] * scale[r] + rs;
      m[r] = mnew;
    }
    for (int t = 0; t < 4; ++t)
      for (int r = 0; r < 8; ++r) Ov[t][r] *= scale[r];
    // P (C/D layout) -> LDS -> A-layout fragment
    for (int r = 0; r < 8; ++r) {
      int row = r + 8 * halfsel;
      Ps[row * 32 + l16]      = f2bf(p0[r]);
      Ps[row * 32 + 16 + l16] = f2bf(p1[r]);
    }
    BFrag ap_;
    for (int p = 0; p < 8; ++p) {
      int kd = ((p & 4) << 2) + halfsel * 8 + ((p & 3) << 1);
      ap_.u[p] = *(const unsigned int*)(Ps + l16 * 32 + kd);
    }
    // PV: B-fragments straight from Vt[d][key] (key contiguous)
    for (int t = 0; t < 4; ++t) {
      BFrag bv;
      const unsigned short* vr =
          Vt + headoff + (size_t)(t * 16 + l16) * T_DIM + kstart + halfsel * 8;
      for (int p = 0; p < 8; ++p)
        bv.u[p] = *(const unsigned int*)(vr + ((p & 4) << 2) + ((p & 3) << 1));
      Ov[t] = wmma_bf16(ap_, bv, Ov[t]);
    }
  }
  // normalize + store Y[b][t][h*64+d]
  for (int r = 0; r < 8; ++r) {
    float inv = 1.0f / lsum[r];
    int trow = qbase + r + 8 * halfsel;
    size_t base = ((size_t)(b * T_DIM + trow)) * C_DIM + h * HD_DIM;
    for (int t = 0; t < 4; ++t)
      Y[base + t * 16 + l16] = f2bf(Ov[t][r] * inv);
  }
}

// ---------------------------------------------------------------------------
// Host launcher
// ---------------------------------------------------------------------------
extern "C" void kernel_launch(void* const* d_in, const int* in_sizes, int n_in,
                              void* d_out, int out_size, void* d_ws, size_t ws_size,
                              hipStream_t stream)
{
  const int*   idx  = (const int*)d_in[0];
  const float* cosb = (const float*)d_in[1];
  const float* sinb = (const float*)d_in[2];
  const float* wte  = (const float*)d_in[3];
  const float* vemb = (const float*)d_in[4];
  const float* wdn  = (const float*)d_in[5];
  const float* wukv = (const float*)d_in[6];
  const float* wuq  = (const float*)d_in[7];
  const float* vg   = (const float*)d_in[8];
  const float* apj  = (const float*)d_in[9];
  const float* fcw  = (const float*)d_in[10];
  const float* pjw  = (const float*)d_in[11];
  const float* rl   = (const float*)d_in[12];
  const float* xl   = (const float*)d_in[13];
  const float* lmh  = (const float*)d_in[14];
  float* out = (float*)d_out;

  size_t off = 0;
  auto alloc = [&](size_t bytes) -> void* {
    void* r = (char*)d_ws + off;
    off += (bytes + 255) & ~(size_t)255;
    return r;
  };
  float*          x     = (float*)alloc((size_t)BT_DIM * C_DIM * 4);
  float*          x0    = (float*)alloc((size_t)BT_DIM * C_DIM * 4);
  unsigned short* xnbf  = (unsigned short*)alloc((size_t)BT_DIM * C_DIM * 2);
  float*          downf = (float*)alloc((size_t)BT_DIM * DDOWN * 4);
  unsigned short* ckv   = (unsigned short*)alloc((size_t)BT_DIM * 256 * 2);
  unsigned short* cq    = (unsigned short*)alloc((size_t)BT_DIM * 256 * 2);
  float*          kvf   = (float*)alloc((size_t)BT_DIM * DKV * 4);
  float*          qf    = (float*)alloc((size_t)BT_DIM * C_DIM * 4);
  unsigned short* Qbf   = (unsigned short*)alloc((size_t)BT_DIM * C_DIM * 2);
  unsigned short* Kbf   = (unsigned short*)alloc((size_t)BT_DIM * C_DIM * 2);
  unsigned short* Vbf   = (unsigned short*)alloc((size_t)BT_DIM * C_DIM * 2);
  unsigned short* Ybf   = (unsigned short*)alloc((size_t)BT_DIM * C_DIM * 2);
  unsigned short* hbf   = (unsigned short*)alloc((size_t)BT_DIM * 4096 * 2);

  auto ggrid = [](int N) { return dim3((unsigned)((N + BN - 1) / BN), BT_DIM / BM, 1); };

  embed_norm<<<BT_DIM, 256, 0, stream>>>(idx, wte, x, x0);

  for (int l = 0; l < 4; ++l) {
    resid_norm<<<BT_DIM, 256, 0, stream>>>(x, x0, rl, xl, l, x, xnbf);
    gemm_bf16_wmma<<<ggrid(DDOWN), 256, 0, stream>>>(
        xnbf, wdn + (size_t)l * C_DIM * DDOWN, downf, nullptr,
        BT_DIM, DDOWN, C_DIM, EPI_F32);
    split_down<<<(BT_DIM * 512) / 256, 256, 0, stream>>>(downf, ckv, cq);
    gemm_bf16_wmma<<<ggrid(DKV), 256, 0, stream>>>(
        ckv, wukv + (size_t)l * 256 * DKV, kvf, nullptr,
        BT_DIM, DKV, 256, EPI_F32);
    gemm_bf16_wmma<<<ggrid(C_DIM), 256, 0, stream>>>(
        cq, wuq + (size_t)l * 256 * C_DIM, qf, nullptr,
        BT_DIM, C_DIM, 256, EPI_F32);
    prep_qkv<<<(BT_DIM * H_DIM + 255) / 256, 256, 0, stream>>>(
        downf, kvf, qf, xnbf, cosb, sinb,
        vg + (size_t)l * 32 * H_DIM, vemb + (size_t)l * 32000 * C_DIM, idx,
        Qbf, Kbf, Vbf);
    attn_flash<<<(B_DIM * H_DIM * (T_DIM / 16)) / 8, 256, 0, stream>>>(Qbf, Kbf, Vbf, Ybf);
    gemm_bf16_wmma<<<ggrid(C_DIM), 256, 0, stream>>>(
        Ybf, apj + (size_t)l * C_DIM * C_DIM, x, x,
        BT_DIM, C_DIM, C_DIM, EPI_RESID);
    resid_norm<<<BT_DIM, 256, 0, stream>>>(x, x0, rl, xl, -1, x, xnbf);
    gemm_bf16_wmma<<<ggrid(4096), 256, 0, stream>>>(
        xnbf, fcw + (size_t)l * C_DIM * 4096, hbf, nullptr,
        BT_DIM, 4096, C_DIM, EPI_RELU2_BF16);
    gemm_bf16_wmma<<<ggrid(C_DIM), 256, 0, stream>>>(
        hbf, pjw + (size_t)l * 4096 * C_DIM, x, x,
        BT_DIM, C_DIM, 4096, EPI_RESID);
  }

  resid_norm<<<BT_DIM, 256, 0, stream>>>(x, x0, rl, xl, -1, x, xnbf);
  gemm_bf16_wmma<<<ggrid(32000), 256, 0, stream>>>(
      xnbf, lmh, out, nullptr, BT_DIM, 32000, C_DIM, EPI_SOFTCAP);
}
